// AttentionwithRoPEBase_61065845015353
// MI455X (gfx1250) — matrix operand order repaired and checked by reading, not compiled, not run on hardware
//
#include <hip/hip_runtime.h>

#define B_  2
#define N_  2048
#define D_  1024
#define H_  16
#define HD_ 64
#define LDK 40   // kernel1/3 LDS tile stride in shorts (80 B, 16 B aligned)

typedef __attribute__((ext_vector_type(16))) __bf16 v16bf;
typedef __attribute__((ext_vector_type(2)))  __bf16 v2bf;
typedef __attribute__((ext_vector_type(8)))  float  v8f;
typedef unsigned int u32;
typedef int v4i __attribute__((vector_size(16)));   // pointee type expected by async builtins

#define AS1 __attribute__((address_space(1)))
#define AS3 __attribute__((address_space(3)))

#if __has_builtin(__builtin_amdgcn_global_load_async_to_lds_b128) && \
    __has_builtin(__builtin_amdgcn_s_wait_asynccnt)
#define HAVE_ASYNC_LDS 1
#define ASYNC_B128(gsrc, ldst) \
  __builtin_amdgcn_global_load_async_to_lds_b128((AS1 v4i*)(gsrc), (AS3 v4i*)(ldst), 0, 0)
#else
#define HAVE_ASYNC_LDS 0
#endif

// ---------- bf16 helpers ----------
__device__ __forceinline__ unsigned short f2us_bf16_sw(float f) {
  union { float f; unsigned u; } v; v.f = f;
  unsigned r = v.u + 0x7FFFu + ((v.u >> 16) & 1u);
  return (unsigned short)(r >> 16);
}

#if __has_builtin(__builtin_amdgcn_cvt_pk_bf16_f32)
__device__ __forceinline__ u32 pack_bf16x2(float lo, float hi) {
  v2bf p = __builtin_amdgcn_cvt_pk_bf16_f32(lo, hi);   // v_cvt_pk_bf16_f32
  u32 r; __builtin_memcpy(&r, &p, 4); return r;
}
__device__ __forceinline__ unsigned short f2us_bf16(float f) {
  v2bf p = __builtin_amdgcn_cvt_pk_bf16_f32(f, 0.f);
  unsigned short r; __builtin_memcpy(&r, &p, 2); return r;
}
#else
__device__ __forceinline__ u32 pack_bf16x2(float lo, float hi) {
  return (u32)f2us_bf16_sw(lo) | ((u32)f2us_bf16_sw(hi) << 16);
}
__device__ __forceinline__ unsigned short f2us_bf16(float f) { return f2us_bf16_sw(f); }
#endif

// ---------- WMMA fragment builders (wave32, contiguous k = 16h + s order in A and B) ----------
// A 16x32 bf16: lanes 0-15 -> M=0..15 (K 0-15), lanes 16-31 same M
// (K 16-31).  Contiguous 16 B runs -> ds_load_b128.
__device__ __forceinline__ v16bf make_a_frag(const unsigned short* tile, int stride, int k0) {
  const int lane = threadIdx.x;
  const int m  = lane & 15;
  const int kb = (lane >> 4) << 4;
  const unsigned short* row = (const unsigned short*)
      __builtin_assume_aligned(tile + m * stride + k0 + kb, 16);
  union { v16bf v; unsigned short s[16]; } f;
  __builtin_memcpy(&f.s[0], row,      16);
  __builtin_memcpy(&f.s[8], row + 8,  16);
  return f.v;
}

// B 32x16 bf16, tile stored "N-major" (tile[n][k] == B[k][n]):
// lanes 0-15 -> K=0..15, lanes 16-31 -> K=16..31, lane%16 = N.
__device__ __forceinline__ v16bf make_b_nt(const unsigned short* tile, int stride, int k0) {
  const int lane = threadIdx.x;
  const int n  = lane & 15;
  const int kb = (lane >> 4) << 4;
  const unsigned short* row = (const unsigned short*)
      __builtin_assume_aligned(tile + n * stride + k0 + kb, 16);
  union { v16bf v; unsigned short s[16]; } f;
  __builtin_memcpy(f.s, row, 32);
  return f.v;
}

// B 32x16 bf16, tile stored "K-major" (tile[k][n] == B[k][n]).
__device__ __forceinline__ v16bf make_b_t(const unsigned short* tile, int stride, int k0) {
  const int lane = threadIdx.x;
  const int n  = lane & 15;
  const int kb = (lane >> 4) << 4;
  const unsigned short* col = tile + (k0 + kb) * stride + n;
  v16bf f;
#pragma unroll
  for (int j = 0; j < 16; ++j) {
    unsigned short u = col[j * stride];
    __bf16 b; __builtin_memcpy(&b, &u, 2);
    f[j] = b;
  }
  return f;
}

__device__ __forceinline__ v8f wmma_bf16(v16bf a, v16bf b, v8f c) {
  return __builtin_amdgcn_wmma_f32_16x16x32_bf16(false, a, false, b, (short)0, c, false, false);
}

// ==========================================================================
// Kernel 1: fused QKV projection (+bias, +RoPE for Q/K), bf16 output in
// head-major [B,H,N,HD].  gridDim.z selects {Q,K,V}.
// 256 threads = 8 waves; block tile = 64 rows x 64 cols (one head).
// ==========================================================================
__global__ __launch_bounds__(256)
void qkv_rope_kernel(const float* __restrict__ x,
                     const float* __restrict__ Wq, const float* __restrict__ Wk,
                     const float* __restrict__ Wv,
                     const float* __restrict__ bq, const float* __restrict__ bk,
                     const float* __restrict__ bv,
                     unsigned short* __restrict__ Qbuf,
                     unsigned short* __restrict__ Kbuf,
                     unsigned short* __restrict__ Vbuf) {
  __shared__ unsigned short aS[64 * LDK];
  __shared__ unsigned short bS[64 * LDK];
  __shared__ float          rS[64 * 65];

  const int z = blockIdx.z;
  const float* W    = (z == 0) ? Wq : (z == 1) ? Wk : Wv;
  const float* bias = (z == 0) ? bq : (z == 1) ? bk : bv;
  unsigned short* dst = (z == 0) ? Qbuf : (z == 1) ? Kbuf : Vbuf;

  const int row0 = blockIdx.y * 64;
  const int col0 = blockIdx.x * 64;          // head-aligned (64 == HD)
  const int tid  = threadIdx.y * 32 + threadIdx.x;
  const int wid  = threadIdx.y;
  const int lane = threadIdx.x;
  const int mt = (wid & 3) * 16;
  const int nt = (wid >> 2) * 32;

  v8f acc0 = {}, acc1 = {};

  for (int kk = 0; kk < D_; kk += 32) {
    // 64x32 f32 tile = 512 float4 chunks; b128 loads + packed bf16x2 stores
#pragma unroll
    for (int i = 0; i < 2; ++i) {
      const int ch = tid + i * 256;
      const int r  = ch >> 3;
      const int c4 = (ch & 7) << 2;
      const float4 va = *(const float4*)(x + (size_t)(row0 + r) * D_ + kk + c4);
      const float4 vb = *(const float4*)(W + (size_t)(col0 + r) * D_ + kk + c4);
      uint2 pa, pb;
      pa.x = pack_bf16x2(va.x, va.y); pa.y = pack_bf16x2(va.z, va.w);
      pb.x = pack_bf16x2(vb.x, vb.y); pb.y = pack_bf16x2(vb.z, vb.w);
      *(uint2*)(aS + r * LDK + c4) = pa;
      *(uint2*)(bS + r * LDK + c4) = pb;
    }
    if (kk + 32 < D_) {                      // gfx1250 global_prefetch_b8
      __builtin_prefetch(&x[(size_t)(row0 + (tid >> 3)) * D_ + kk + 32 + ((tid & 7) << 2)], 0, 1);
      __builtin_prefetch(&W[(size_t)(col0 + (tid >> 3)) * D_ + kk + 32 + ((tid & 7) << 2)], 0, 1);
    }
    __syncthreads();

    const v16bf af  = make_a_frag(aS + mt * LDK, LDK, 0);
    const v16bf bf0 = make_b_nt(bS + nt * LDK, LDK, 0);
    const v16bf bf1 = make_b_nt(bS + (nt + 16) * LDK, LDK, 0);
    acc0 = wmma_bf16(af, bf0, acc0);
    acc1 = wmma_bf16(af, bf1, acc1);
    __syncthreads();
  }

  // C/D layout -> f32 LDS tile (+bias)
  {
    const int n  = lane & 15;
    const int mb = (lane >> 4) << 3;
#pragma unroll
    for (int r = 0; r < 8; ++r) {
      const int m = mt + mb + r;
      rS[m * 65 + nt + n]      = acc0[r] + bias[col0 + nt + n];
      rS[m * 65 + nt + 16 + n] = acc1[r] + bias[col0 + nt + 16 + n];
    }
  }
  __syncthreads();

  // RoPE (Q,K) + bf16x2 packed store, head-major layout
  const int h = col0 >> 6;
#pragma unroll
  for (int i = 0; i < 8; ++i) {              // 2048 bf16-pairs / 256 threads
    const int idx = tid + i * 256;
    const int r  = idx >> 5;
    const int c2 = (idx & 31) << 1;          // even column
    const int row = row0 + r;
    const int b = row >> 11;
    const int n = row & (N_ - 1);
    float v0, v1;
    if (z < 2) {
      const int p0 = c2 & 31, p1 = (c2 + 1) & 31;
      const float if0 = __expf(-(float)p0 * 0.28782313662425572f);  // ln(1e4)/32
      const float if1 = __expf(-(float)p1 * 0.28782313662425572f);
      float s0, co0, s1, co1;
      __sincosf((float)n * if0, &s0, &co0);
      __sincosf((float)n * if1, &s1, &co1);
      const float a0 = rS[r * 65 + p0],      a1 = rS[r * 65 + p1];
      const float b0 = rS[r * 65 + p0 + 32], b1 = rS[r * 65 + p1 + 32];
      if (c2 < 32) { v0 = a0 * co0 - b0 * s0; v1 = a1 * co1 - b1 * s1; }
      else         { v0 = b0 * co0 + a0 * s0; v1 = b1 * co1 + a1 * s1; }
    } else {
      v0 = rS[r * 65 + c2]; v1 = rS[r * 65 + c2 + 1];
    }
    *(u32*)(dst + (((size_t)b * H_ + h) * N_ + n) * HD_ + c2) = pack_bf16x2(v0, v1);
  }
}

// ==========================================================================
// Kernel 2: flash attention per (b,h).  8 waves x 16 full query rows each:
// softmax row stats reduce across the 16 lanes of a half-wave (matches the
// WMMA C/D layout).  K/V tiles staged with gfx1250 async LDS loads.
// ==========================================================================
__global__ __launch_bounds__(256)
void flash_attn_kernel(const unsigned short* __restrict__ Qbuf,
                       const unsigned short* __restrict__ Kbuf,
                       const unsigned short* __restrict__ Vbuf,
                       unsigned short* __restrict__ Obuf) {
  __shared__ unsigned short kS[64 * 72];
  __shared__ unsigned short vS[64 * 72];
  __shared__ unsigned short pS[8 * 16 * 72];

  const int bh = blockIdx.y;
  const int b  = bh >> 4;
  const int h  = bh & 15;
  const int q0 = blockIdx.x * 128;
  const int tid  = threadIdx.y * 32 + threadIdx.x;
  const int wid  = threadIdx.y;
  const int lane = threadIdx.x;

  const unsigned short* Qbase = Qbuf + (size_t)bh * N_ * HD_;
  const unsigned short* Kbase = Kbuf + (size_t)bh * N_ * HD_;
  const unsigned short* Vbase = Vbuf + (size_t)bh * N_ * HD_;

  // Q fragments for this wave's 16 rows, 16 B chunks straight into A layout
  v16bf aq0, aq1;
  {
    const int m  = lane & 15;
    const int kb = (lane >> 4) << 4;
    const unsigned short* qrow = Qbase + (size_t)(q0 + wid * 16 + m) * HD_;
    union { v16bf v; unsigned short s[16]; } f0, f1;
    __builtin_memcpy(&f0.s[0], qrow + kb,           16);
    __builtin_memcpy(&f0.s[8], qrow + kb + 8,       16);
    __builtin_memcpy(&f1.s[0], qrow + 32 + kb,      16);
    __builtin_memcpy(&f1.s[8], qrow + 32 + kb + 8,  16);
    aq0 = f0.v; aq1 = f1.v;
  }

  v8f o0 = {}, o1 = {}, o2 = {}, o3 = {};
  float mrow[8], lrow[8];
#pragma unroll
  for (int r = 0; r < 8; ++r) { mrow[r] = -1e30f; lrow[r] = 0.f; }

  unsigned short* pW = pS + wid * 16 * 72;
  const float scale = 0.125f;                // HD^-0.5

  for (int t = 0; t < N_; t += 64) {
    // --- stage K,V tiles: 512 x 16 B chunks each ---
#if HAVE_ASYNC_LDS
#pragma unroll
    for (int i = 0; i < 2; ++i) {
      const int ch = tid + i * 256;
      const int r  = ch >> 3;
      const int c8 = (ch & 7) << 3;
      ASYNC_B128(Kbase + (size_t)(t + r) * HD_ + c8, kS + r * 72 + c8);
      ASYNC_B128(Vbase + (size_t)(t + r) * HD_ + c8, vS + r * 72 + c8);
    }
    __builtin_amdgcn_s_wait_asynccnt(0);
#else
#pragma unroll
    for (int i = 0; i < 2; ++i) {
      const int ch = tid + i * 256;
      const int r  = ch >> 3;
      const int c8 = (ch & 7) << 3;
      *(uint4*)(kS + r * 72 + c8) = *(const uint4*)(Kbase + (size_t)(t + r) * HD_ + c8);
      *(uint4*)(vS + r * 72 + c8) = *(const uint4*)(Vbase + (size_t)(t + r) * HD_ + c8);
    }
#endif
    __syncthreads();

    // S = Q K^T (16 x 64 per wave)
    v8f s0 = {}, s1 = {}, s2 = {}, s3 = {};
    {
      v16bf b0 = make_b_nt(kS,           72, 0);
      v16bf b1 = make_b_nt(kS + 16 * 72, 72, 0);
      v16bf b2 = make_b_nt(kS + 32 * 72, 72, 0);
      v16bf b3 = make_b_nt(kS + 48 * 72, 72, 0);
      s0 = wmma_bf16(aq0, b0, s0); s1 = wmma_bf16(aq0, b1, s1);
      s2 = wmma_bf16(aq0, b2, s2); s3 = wmma_bf16(aq0, b3, s3);
      b0 = make_b_nt(kS,           72, 32);
      b1 = make_b_nt(kS + 16 * 72, 72, 32);
      b2 = make_b_nt(kS + 32 * 72, 72, 32);
      b3 = make_b_nt(kS + 48 * 72, 72, 32);
      s0 = wmma_bf16(aq1, b0, s0); s1 = wmma_bf16(aq1, b1, s1);
      s2 = wmma_bf16(aq1, b2, s2); s3 = wmma_bf16(aq1, b3, s3);
    }

    // online softmax; each lane: 8 rows x 4 cols; width-16 xor row reduction
    float p0[8], p1[8], p2[8], p3[8];
#pragma unroll
    for (int r = 0; r < 8; ++r) {
      const float v0 = s0[r] * scale, v1 = s1[r] * scale;
      const float v2 = s2[r] * scale, v3 = s3[r] * scale;
      float mx = fmaxf(fmaxf(v0, v1), fmaxf(v2, v3));
#pragma unroll
      for (int off = 8; off >= 1; off >>= 1)
        mx = fmaxf(mx, __shfl_xor(mx, off, 16));
      const float mnew  = fmaxf(mrow[r], mx);
      const float alpha = __expf(mrow[r] - mnew);
      const float e0 = __expf(v0 - mnew), e1 = __expf(v1 - mnew);
      const float e2 = __expf(v2 - mnew), e3 = __expf(v3 - mnew);
      float sum = e0 + e1 + e2 + e3;
#pragma unroll
      for (int off = 8; off >= 1; off >>= 1)
        sum += __shfl_xor(sum, off, 16);
      lrow[r] = lrow[r] * alpha + sum;
      mrow[r] = mnew;
      o0[r] *= alpha; o1[r] *= alpha; o2[r] *= alpha; o3[r] *= alpha;
      p0[r] = e0; p1[r] = e1; p2[r] = e2; p3[r] = e3;
    }

    // C-layout P -> per-wave LDS strip (bf16) -> A-layout fragments
    {
      const int n  = lane & 15;
      const int mb = (lane >> 4) << 3;
#pragma unroll
      for (int r = 0; r < 8; ++r) {
        unsigned short* prow = pW + (mb + r) * 72;
        prow[n]      = f2us_bf16(p0[r]);
        prow[16 + n] = f2us_bf16(p1[r]);
        prow[32 + n] = f2us_bf16(p2[r]);
        prow[48 + n] = f2us_bf16(p3[r]);
      }
    }
    // O += P @ V
    {
      const v16bf ap0 = make_a_frag(pW, 72, 0);
      const v16bf ap1 = make_a_frag(pW, 72, 32);
      v16bf b0 = make_b_t(vS,      72, 0);
      v16bf b1 = make_b_t(vS + 16, 72, 0);
      v16bf b2 = make_b_t(vS + 32, 72, 0);
      v16bf b3 = make_b_t(vS + 48, 72, 0);
      o0 = wmma_bf16(ap0, b0, o0); o1 = wmma_bf16(ap0, b1, o1);
      o2 = wmma_bf16(ap0, b2, o2); o3 = wmma_bf16(ap0, b3, o3);
      b0 = make_b_t(vS,      72, 32);
      b1 = make_b_t(vS + 16, 72, 32);
      b2 = make_b_t(vS + 32, 72, 32);
      b3 = make_b_t(vS + 48, 72, 32);
      o0 = wmma_bf16(ap1, b0, o0); o1 = wmma_bf16(ap1, b1, o1);
      o2 = wmma_bf16(ap1, b2, o2); o3 = wmma_bf16(ap1, b3, o3);
    }
    __syncthreads();
  }

  // normalize and store O as bf16 in [B, N, D] (heads re-interleaved)
  {
    const int n  = lane & 15;
    const int mb = (lane >> 4) << 3;
#pragma unroll
    for (int r = 0; r < 8; ++r) {
      const float inv_l = 1.0f / lrow[r];
      const size_t grow = (size_t)b * N_ + q0 + wid * 16 + mb + r;
      unsigned short* orow = Obuf + grow * D_ + h * HD_;
      orow[n]      = f2us_bf16(o0[r] * inv_l);
      orow[16 + n] = f2us_bf16(o1[r] * inv_l);
      orow[32 + n] = f2us_bf16(o2[r] * inv_l);
      orow[48 + n] = f2us_bf16(o3[r] * inv_l);
    }
  }
}

// ==========================================================================
// Kernel 3: out = O @ Wo^T + bo (bf16 in, f32 out); O tile via async LDS
// ==========================================================================
__global__ __launch_bounds__(256)
void out_proj_kernel(const unsigned short* __restrict__ Obuf,
                     const float* __restrict__ Wo, const float* __restrict__ bo,
                     float* __restrict__ out) {
  __shared__ unsigned short aS[64 * LDK];
  __shared__ unsigned short bS[64 * LDK];

  const int row0 = blockIdx.y * 64;
  const int col0 = blockIdx.x * 64;
  const int tid  = threadIdx.y * 32 + threadIdx.x;
  const int wid  = threadIdx.y;
  const int lane = threadIdx.x;
  const int mt = (wid & 3) * 16;
  const int nt = (wid >> 2) * 32;

  v8f acc0 = {}, acc1 = {};
  for (int kk = 0; kk < D_; kk += 32) {
    // A tile: bf16 -> bf16, 256 x 16 B chunks, async straight to LDS
    {
      const int r  = tid >> 2;
      const int c8 = (tid & 3) << 3;
#if HAVE_ASYNC_LDS
      ASYNC_B128(Obuf + (size_t)(row0 + r) * D_ + kk + c8, aS + r * LDK + c8);
#else
      *(uint4*)(aS + r * LDK + c8) =
          *(const uint4*)(Obuf + (size_t)(row0 + r) * D_ + kk + c8);
#endif
    }
    // B tile: f32 -> bf16
#pragma unroll
    for (int i = 0; i < 2; ++i) {
      const int ch = tid + i * 256;
      const int r  = ch >> 3;
      const int c4 = (ch & 7) << 2;
      const float4 vb = *(const float4*)(Wo + (size_t)(col0 + r) * D_ + kk + c4);
      uint2 pb; pb.x = pack_bf16x2(vb.x, vb.y); pb.y = pack_bf16x2(vb.z, vb.w);
      *(uint2*)(bS + r * LDK + c4) = pb;
    }
    if (kk + 32 < D_) {
      __builtin_prefetch(&Wo[(size_t)(col0 + (tid >> 3)) * D_ + kk + 32 + ((tid & 7) << 2)], 0, 1);
    }
#if HAVE_ASYNC_LDS
    __builtin_amdgcn_s_wait_asynccnt(0);
#endif
    __syncthreads();

    const v16bf af  = make_a_frag(aS + mt * LDK, LDK, 0);
    const v16bf bf0 = make_b_nt(bS + nt * LDK, LDK, 0);
    const v16bf bf1 = make_b_nt(bS + (nt + 16) * LDK, LDK, 0);
    acc0 = wmma_bf16(af, bf0, acc0);
    acc1 = wmma_bf16(af, bf1, acc1);
    __syncthreads();
  }
  {
    const int n  = lane & 15;
    const int mb = (lane >> 4) << 3;
#pragma unroll
    for (int r = 0; r < 8; ++r) {
      const size_t m = (size_t)row0 + mt + mb + r;
      out[m * D_ + col0 + nt + n]      = acc0[r] + bo[col0 + nt + n];
      out[m * D_ + col0 + nt + 16 + n] = acc1[r] + bo[col0 + nt + 16 + n];
    }
  }
}

// ==========================================================================
extern "C" void kernel_launch(void* const* d_in, const int* in_sizes, int n_in,
                              void* d_out, int out_size, void* d_ws, size_t ws_size,
                              hipStream_t stream) {
  const float* x  = (const float*)d_in[0];
  const float* Wq = (const float*)d_in[1];
  const float* Wk = (const float*)d_in[2];
  const float* Wv = (const float*)d_in[3];
  const float* Wo = (const float*)d_in[4];
  const float* bq = (const float*)d_in[5];
  const float* bk = (const float*)d_in[6];
  const float* bv = (const float*)d_in[7];
  const float* bo = (const float*)d_in[8];
  float* out = (float*)d_out;

  const size_t headElems = (size_t)B_ * H_ * N_ * HD_;   // 4M elems
  unsigned short* Qbuf = (unsigned short*)d_ws;
  unsigned short* Kbuf = Qbuf + headElems;
  unsigned short* Vbuf = Kbuf + headElems;
  unsigned short* Obuf = Vbuf + headElems;               // 32 MB total ws

  dim3 blk(32, 8);
  qkv_rope_kernel<<<dim3(D_ / 64, (B_ * N_) / 64, 3), blk, 0, stream>>>(
      x, Wq, Wk, Wv, bq, bk, bv, Qbuf, Kbuf, Vbuf);
  flash_attn_kernel<<<dim3(N_ / 128, B_ * H_), blk, 0, stream>>>(
      Qbuf, Kbuf, Vbuf, Obuf);
  out_proj_kernel<<<dim3(D_ / 64, (B_ * N_) / 64), blk, 0, stream>>>(
      Obuf, Wo, bo, out);
}
